// SlotAttention_49770081026757
// MI455X (gfx1250) — matrix operand order repaired and checked by reading, not compiled, not run on hardware
//
#include <hip/hip_runtime.h>

// ---------------------------------------------------------------------------
// Slot Attention with MESH/Sinkhorn for MI455X (gfx1250, wave32, WMMA).
// GEMMs (projections, cdist inner product, attn@v) use v_wmma_f32_16x16x32_f16.
// v and attn are kept transposed in f16 so all WMMA fragments load as
// contiguous b128 vectors. Sinkhorn/MESH sweeps are L2-resident f32 kernels.
// ---------------------------------------------------------------------------

#define Bm 16
#define Nn 16384
#define Dm 64
#define Sm 7
#define SP 8            // S padded for coalescing
#define Hm 128
#define NT (Bm * Nn)    // 262144 rows
#define EPSc 1e-8f
#define MESH_LRc 5.0f

typedef __attribute__((ext_vector_type(16))) _Float16 v16h;
typedef __attribute__((ext_vector_type(8)))  _Float16 v8h;
typedef __attribute__((ext_vector_type(8)))  float    v8f;
typedef _Float16 h16;

// A-matrix (16x32 f16): lane holds K = hi*16 + h, h=0..15 -> one 32B run
// (same K order as the B fragment).
__device__ __forceinline__ v16h load_a_frag(const h16* rowp, int hi) {
  return *(const v16h*)(rowp + hi * 16);
}
// B-matrix (32x16 f16): lane holds K = hi*16 + h, h=0..15 -> one 32B run.
__device__ __forceinline__ v16h load_b_frag(const h16* colp, int hi) {
  return *(const v16h*)(colp + hi * 16);
}

// ---------------------------------------------------------------------------
__global__ void k_init_slots(const float* __restrict__ si, const float* __restrict__ mu,
                             const float* __restrict__ ls, float* __restrict__ slots) {
  int i = blockIdx.x * blockDim.x + threadIdx.x;
  if (i >= Bm * Sm * Dm) return;
  int d = i % Dm;
  int s = (i / Dm) % Sm;
  int b = i / (Dm * Sm);
  slots[((size_t)b * SP + s) * Dm + d] = mu[d] + expf(ls[d]) * si[i];
}

// ---------------------------------------------------------------------------
// LayerNorm(inputs) -> f16 in LDS; k = xn@Wk^T (row-major), v = xn@Wv^T
// (transposed vhT[b][d][n], single b128 store/lane); wi logit.
__global__ __launch_bounds__(256) void k_ln_proj(
    const float* __restrict__ x, const float* __restrict__ Wk, const float* __restrict__ Wv,
    const float* __restrict__ g, const float* __restrict__ bta,
    const float* __restrict__ wiW, const float* __restrict__ wib,
    h16* __restrict__ kh, h16* __restrict__ vhT, float* __restrict__ wlog) {
  __shared__ h16 xs[128][Dm];     // 16 KB
  __shared__ h16 wks[Dm * Dm];    // 8 KB
  __shared__ h16 wvs[Dm * Dm];    // 8 KB
  int tid = threadIdx.x;
  int row0 = blockIdx.x * 128;
  int bb = row0 >> 14;            // batch (blocks never straddle a batch)
  int n0 = row0 & (Nn - 1);

  for (int i = tid; i < Dm * Dm; i += 256) {
    wks[i] = (h16)Wk[i];
    wvs[i] = (h16)Wv[i];
  }

  // LayerNorm: 2 threads per row (32 elems each), combine via shfl_xor(1)
  int r = tid >> 1, half = tid & 1;
  const float* xr = x + (size_t)(row0 + r) * Dm + half * 32;
  float vals[32];
  float s1 = 0.f, s2 = 0.f;
#pragma unroll
  for (int j = 0; j < 32; j++) {
    float v = xr[j];
    vals[j] = v;
    s1 += v;
    s2 += v * v;
  }
  s1 += __shfl_xor(s1, 1, 32);
  s2 += __shfl_xor(s2, 1, 32);
  float mean = s1 * (1.0f / Dm);
  float var = s2 * (1.0f / Dm) - mean * mean;
  float inv = rsqrtf(var + 1e-5f);
  float dot = 0.f;
#pragma unroll
  for (int j = 0; j < 32; j++) {
    int c = half * 32 + j;
    float xn = (vals[j] - mean) * inv * g[c] + bta[c];
    xs[r][c] = (h16)xn;
    dot += xn * wiW[c];
  }
  dot += __shfl_xor(dot, 1, 32);
  if (half == 0) wlog[row0 + r] = dot + wib[0];
  __syncthreads();

  // WMMA: each wave one 16-row tile, 4 output column tiles per weight matrix
  int lane = tid & 31, wave = tid >> 5;
  int m = lane & 15, hi = lane >> 4;
  int rt = wave * 16;
  v16h A0 = load_a_frag(&xs[rt + m][0], hi);
  v16h A1 = load_a_frag(&xs[rt + m][32], hi);
  int ncol = lane & 15;
  for (int w = 0; w < 2; w++) {
    const h16* wsrc = w ? wvs : wks;
    for (int nt = 0; nt < 4; nt++) {
      int d = nt * 16 + ncol;
      v16h B0 = load_b_frag(wsrc + d * Dm, hi);        // B[k=e][n=d] = W[d][e]
      v16h B1 = load_b_frag(wsrc + d * Dm + 32, hi);
      v8f c = {};
      c = __builtin_amdgcn_wmma_f32_16x16x32_f16(false, A0, false, B0, (short)0, c, false, false);
      c = __builtin_amdgcn_wmma_f32_16x16x32_f16(false, A1, false, B1, (short)0, c, false, false);
      if (w == 0) {
        // k row-major (cdist A-fragment wants K-contiguous rows)
#pragma unroll
        for (int rr = 0; rr < 8; rr++) {
          int M = rr + (hi ? 8 : 0);
          kh[(size_t)(row0 + rt + M) * Dm + d] = (h16)c[rr];
        }
      } else {
        // v transposed: lane holds 8 consecutive rows of one column d
        v8h hv;
#pragma unroll
        for (int rr = 0; rr < 8; rr++) hv[rr] = (h16)c[rr];
        *(v8h*)(vhT + ((size_t)(bb * Dm + d)) * Nn + n0 + rt + (hi ? 8 : 0)) = hv;
      }
    }
  }
}

// ---------------------------------------------------------------------------
__global__ void k_k2(const h16* __restrict__ kh, float* __restrict__ k2) {
  int i = blockIdx.x * 256 + threadIdx.x;  // row id < NT
  const h16* p = kh + (size_t)i * Dm;
  float s = 0.f;
#pragma unroll
  for (int j = 0; j < Dm; j++) {
    float v = (float)p[j];
    s += v * v;
  }
  k2[i] = s;
}

// ---------------------------------------------------------------------------
// log_a = log_softmax over N + log(S); online LSE + block tree reduce.
__global__ __launch_bounds__(256) void k_amarg(const float* __restrict__ wlog,
                                               float* __restrict__ log_a) {
  __shared__ float rm[256], rs[256];
  int b = blockIdx.x, tid = threadIdx.x;
  const float* p = wlog + (size_t)b * Nn;
  float m = -1e30f, s = 0.f;
  for (int n = tid; n < Nn; n += 256) {
    float v = p[n];
    if (v > m) { s = s * expf(m - v) + 1.f; m = v; }
    else       { s += expf(v - m); }
  }
  rm[tid] = m; rs[tid] = s;
  __syncthreads();
  for (int off = 128; off > 0; off >>= 1) {
    if (tid < off) {
      float m2 = rm[tid + off], s2 = rs[tid + off];
      float M = fmaxf(rm[tid], m2);
      rs[tid] = rs[tid] * expf(rm[tid] - M) + s2 * expf(m2 - M);
      rm[tid] = M;
    }
    __syncthreads();
  }
  float lse = rm[0] + logf(rs[0]);
  float lgS = logf((float)Sm);
  for (int n = tid; n < Nn; n += 256)
    log_a[(size_t)b * Nn + n] = p[n] - lse + lgS;
}

// ---------------------------------------------------------------------------
// Per batch: LN(slots), slot marginal log_b, q = s_n@Wq^T (f16, padded 16 rows),
// q2, and zero the sinkhorn dual v.
__global__ __launch_bounds__(64) void k_slotsnorm(
    const float* __restrict__ slots, const float* __restrict__ g, const float* __restrict__ bta,
    const float* __restrict__ wsW, const float* __restrict__ wsb, const float* __restrict__ Wq,
    float* __restrict__ log_b, float* __restrict__ q2, h16* __restrict__ qh,
    float* __restrict__ vdual) {
  __shared__ float red[64];
  __shared__ float sn[Sm][Dm];
  __shared__ float wl[Sm];
  int b = blockIdx.x, t = threadIdx.x;
  for (int s = 0; s < Sm; s++) {
    float v = slots[((size_t)b * SP + s) * Dm + t];
    red[t] = v; __syncthreads();
    for (int off = 32; off > 0; off >>= 1) { if (t < off) red[t] += red[t + off]; __syncthreads(); }
    float mean = red[0] * (1.f / Dm); __syncthreads();
    float dv = v - mean;
    red[t] = dv * dv; __syncthreads();
    for (int off = 32; off > 0; off >>= 1) { if (t < off) red[t] += red[t + off]; __syncthreads(); }
    float var = red[0] * (1.f / Dm); __syncthreads();
    float x = dv * rsqrtf(var + 1e-5f) * g[t] + bta[t];
    sn[s][t] = x;
    red[t] = x * wsW[t]; __syncthreads();
    for (int off = 32; off > 0; off >>= 1) { if (t < off) red[t] += red[t + off]; __syncthreads(); }
    if (t == 0) wl[s] = red[0] + wsb[0];
    __syncthreads();
  }
  if (t < Sm) {
    float M = wl[0];
    for (int s = 1; s < Sm; s++) M = fmaxf(M, wl[s]);
    float S = 0.f;
    for (int s = 0; s < Sm; s++) S += expf(wl[s] - M);
    log_b[b * SP + t] = wl[t] - M - logf(S) + logf((float)Sm);
  }
  if (t < SP) vdual[b * SP + t] = 0.f;
  __syncthreads();
  for (int s = 0; s < 16; s++) {
    if (s < Sm) {
      float acc = 0.f;
      for (int e = 0; e < Dm; e++) acc += sn[s][e] * Wq[t * Dm + e];
      qh[((size_t)b * 16 + s) * Dm + t] = (h16)acc;
      red[t] = acc * acc; __syncthreads();
      for (int off = 32; off > 0; off >>= 1) { if (t < off) red[t] += red[t + off]; __syncthreads(); }
      if (t == 0) q2[b * SP + s] = red[0];
      __syncthreads();
    } else {
      qh[((size_t)b * 16 + s) * Dm + t] = (h16)0.f;  // zero padding rows
    }
  }
}

// ---------------------------------------------------------------------------
// C[b,n,s] = sqrt(max(k2 + q2 - 2 k.q, 0) + 1e-12), k.q via WMMA (one wave/16 rows)
__global__ __launch_bounds__(32) void k_cdist(
    const h16* __restrict__ kh, const h16* __restrict__ qh,
    const float* __restrict__ k2, const float* __restrict__ q2, float* __restrict__ C) {
  int blk = blockIdx.x;
  int b = blk >> 10;            // / (Nn/16)
  int row0 = (blk & 1023) * 16;
  int lane = threadIdx.x;
  int m = lane & 15, hi = lane >> 4;
  const h16* kp = kh + ((size_t)b * Nn + row0 + m) * Dm;
  v16h A0 = load_a_frag(kp, hi);
  v16h A1 = load_a_frag(kp + 32, hi);
  int s = lane & 15;
  const h16* qp = qh + ((size_t)b * 16 + s) * Dm;  // B[k=e][n=s] = q[s][e]
  v16h B0 = load_b_frag(qp, hi);
  v16h B1 = load_b_frag(qp + 32, hi);
  v8f c = {};
  c = __builtin_amdgcn_wmma_f32_16x16x32_f16(false, A0, false, B0, (short)0, c, false, false);
  c = __builtin_amdgcn_wmma_f32_16x16x32_f16(false, A1, false, B1, (short)0, c, false, false);
  if (s < Sm) {
    float q2v = q2[b * SP + s];
#pragma unroll
    for (int rr = 0; rr < 8; rr++) {
      int M = rr + (hi ? 8 : 0);
      int row = row0 + M;
      float k2v = k2[(size_t)b * Nn + row];
      float sq = fmaxf(k2v + q2v - 2.f * c[rr], 0.f) + 1e-12f;
      C[((size_t)b * Nn + row) * SP + s] = sqrtf(sq);
    }
  }
}

// ---------------------------------------------------------------------------
// u = log_a - LSE_s(-C + v)
__global__ void k_u(const float* __restrict__ C, const float* __restrict__ vdual,
                    const float* __restrict__ log_a, float* __restrict__ u) {
  int i = blockIdx.x * 256 + threadIdx.x;  // < NT
  int b = i >> 14;
  const float* Cp = C + (size_t)i * SP;
  const float* vp = vdual + b * SP;
  float m = -1e30f, ssum = 0.f;
#pragma unroll
  for (int s = 0; s < Sm; s++) {
    float t = -Cp[s] + vp[s];
    if (t > m) { ssum = ssum * expf(m - t) + 1.f; m = t; }
    else       { ssum += expf(t - m); }
  }
  u[i] = log_a[i] - (m + logf(ssum));
}

// ---------------------------------------------------------------------------
// v[s] = log_b[s] - LSE_n(-C + u): one block per batch, 7 online LSE accs/thread
__global__ __launch_bounds__(256) void k_v(const float* __restrict__ C,
                                           const float* __restrict__ u,
                                           const float* __restrict__ log_b,
                                           float* __restrict__ vdual) {
  __shared__ float rm[256], rs[256];
  int b = blockIdx.x, tid = threadIdx.x;
  float m[Sm], sm[Sm];
#pragma unroll
  for (int s = 0; s < Sm; s++) { m[s] = -1e30f; sm[s] = 0.f; }
  for (int n = tid; n < Nn; n += 256) {
    float uu = u[(size_t)b * Nn + n];
    const float* Cp = C + ((size_t)b * Nn + n) * SP;
#pragma unroll
    for (int s = 0; s < Sm; s++) {
      float t = -Cp[s] + uu;
      if (t > m[s]) { sm[s] = sm[s] * expf(m[s] - t) + 1.f; m[s] = t; }
      else          { sm[s] += expf(t - m[s]); }
    }
  }
  for (int s = 0; s < Sm; s++) {
    rm[tid] = m[s]; rs[tid] = sm[s];
    __syncthreads();
    for (int off = 128; off > 0; off >>= 1) {
      if (tid < off) {
        float m2 = rm[tid + off], s2 = rs[tid + off];
        float M = fmaxf(rm[tid], m2);
        rs[tid] = rs[tid] * expf(rm[tid] - M) + s2 * expf(m2 - M);
        rm[tid] = M;
      }
      __syncthreads();
    }
    if (tid == 0) vdual[b * SP + s] = log_b[b * SP + s] - (rm[0] + logf(rs[0]));
    __syncthreads();
  }
}

// ---------------------------------------------------------------------------
// MESH step, frozen-duals entropy gradient: C -= lr * T*(ln(T+eps)+1)
__global__ void k_meshgrad(float* __restrict__ C, const float* __restrict__ u,
                           const float* __restrict__ vdual) {
  int i = blockIdx.x * 256 + threadIdx.x;
  int b = i >> 14;
  float uu = u[i];
  float* Cp = C + (size_t)i * SP;
  const float* vp = vdual + b * SP;
#pragma unroll
  for (int s = 0; s < Sm; s++) {
    float T = expf(-Cp[s] + uu + vp[s]);
    float gr = T * (logf(T + EPSc) + 1.f);
    Cp[s] = Cp[s] - MESH_LRc * gr;
  }
}

// ---------------------------------------------------------------------------
// attn = exp(-C+u+v): f32 [b,s,n] to d_out; f16 transposed attnT[b,s,n] for WMMA
__global__ void k_attn(const float* __restrict__ C, const float* __restrict__ u,
                       const float* __restrict__ vdual, h16* __restrict__ attnT,
                       float* __restrict__ attn_out) {
  int i = blockIdx.x * 256 + threadIdx.x;
  int b = i >> 14, n = i & (Nn - 1);
  float uu = u[i];
  const float* Cp = C + (size_t)i * SP;
  const float* vp = vdual + b * SP;
#pragma unroll
  for (int s = 0; s < Sm; s++) {
    float T = expf(-Cp[s] + uu + vp[s]);
    attnT[((size_t)(b * SP + s)) * Nn + n] = (h16)T;
    attn_out[((size_t)(b * Sm + s)) * Nn + n] = T;
  }
  attnT[((size_t)(b * SP + Sm)) * Nn + n] = (h16)0.f;  // padding row
}

// ---------------------------------------------------------------------------
// updates[b,s,d] = sum_n attn[b,s,n] * v[b,n,d] via WMMA.
// 8 waves split the K=N range; partial 16x16 f32 accumulators reduced in LDS.
// attnT/vhT layouts make every fragment a contiguous b128 load.
__global__ __launch_bounds__(256) void k_updates(const h16* __restrict__ attnT,
                                                 const h16* __restrict__ vhT,
                                                 float* __restrict__ upd) {
  __shared__ float accs[8][32][8];  // 8 KB
  int b = blockIdx.x >> 2, nt = blockIdx.x & 3;
  int lane = threadIdx.x & 31, wave = threadIdx.x >> 5;
  int m = lane & 15, hi = lane >> 4;
  int d = nt * 16 + (lane & 15);
  const h16* ap = attnT + ((size_t)(b * SP + (m < SP ? m : 0))) * Nn;
  const h16* bp = vhT + ((size_t)(b * Dm + d)) * Nn;
  int kk0 = wave * (Nn / 8), kk1 = kk0 + (Nn / 8);
  v8f c = {};
  for (int kk = kk0; kk < kk1; kk += 32) {
    __builtin_prefetch(bp + kk + 512, 0, 1);
    v16h A = {};
    if (m < SP) A = load_a_frag(ap + kk, hi);  // row 7 is zero padding
    v16h Bf = load_b_frag(bp + kk, hi);
    c = __builtin_amdgcn_wmma_f32_16x16x32_f16(false, A, false, Bf, (short)0, c, false, false);
  }
#pragma unroll
  for (int rr = 0; rr < 8; rr++) accs[wave][lane][rr] = c[rr];
  __syncthreads();
  if (wave == 0) {
#pragma unroll
    for (int rr = 0; rr < 8; rr++) {
      float s = 0.f;
      for (int w2 = 0; w2 < 8; w2++) s += accs[w2][lane][rr];
      int M = rr + (hi ? 8 : 0);
      if (M < Sm) upd[((size_t)b * SP + M) * Dm + d] = s;
    }
  }
}

// ---------------------------------------------------------------------------
// GRU cell + LN + MLP residual per batch (7 slots); 64 threads = 2 waves
__global__ __launch_bounds__(64) void k_gru_ff(
    const float* __restrict__ upd, const float* __restrict__ slots_in,
    const float* __restrict__ Wih, const float* __restrict__ Whh,
    const float* __restrict__ bih, const float* __restrict__ bhh,
    const float* __restrict__ fc1W, const float* __restrict__ fc1b,
    const float* __restrict__ fc2W, const float* __restrict__ fc2b,
    const float* __restrict__ lng, const float* __restrict__ lnb,
    float* __restrict__ slots_out, float* __restrict__ out_slots) {
  __shared__ float xu[Dm], hp[Dm], red[64], ffin[Dm], ff1[Hm];
  int b = blockIdx.x, t = threadIdx.x;
  for (int s = 0; s < Sm; s++) {
    xu[t] = upd[((size_t)b * SP + s) * Dm + t];
    hp[t] = slots_in[((size_t)b * SP + s) * Dm + t];
    __syncthreads();
    float gi[3], gh[3];
#pragma unroll
    for (int p = 0; p < 3; p++) {
      int j = p * Dm + t;
      float a1 = bih[j], a2 = bhh[j];
      for (int e = 0; e < Dm; e++) {
        a1 += xu[e] * Wih[(size_t)j * Dm + e];
        a2 += hp[e] * Whh[(size_t)j * Dm + e];
      }
      gi[p] = a1; gh[p] = a2;
    }
    float r = 1.f / (1.f + expf(-(gi[0] + gh[0])));
    float z = 1.f / (1.f + expf(-(gi[1] + gh[1])));
    float nn = tanhf(gi[2] + r * gh[2]);
    float h = (1.f - z) * nn + z * hp[t];
    // LayerNorm(h)
    red[t] = h; __syncthreads();
    for (int off = 32; off > 0; off >>= 1) { if (t < off) red[t] += red[t + off]; __syncthreads(); }
    float mean = red[0] * (1.f / Dm); __syncthreads();
    float dv = h - mean;
    red[t] = dv * dv; __syncthreads();
    for (int off = 32; off > 0; off >>= 1) { if (t < off) red[t] += red[t + off]; __syncthreads(); }
    float var = red[0] * (1.f / Dm); __syncthreads();
    ffin[t] = dv * rsqrtf(var + 1e-5f) * lng[t] + lnb[t];
    __syncthreads();
#pragma unroll
    for (int p = 0; p < 2; p++) {
      int j = p * Dm + t;
      float a = fc1b[j];
      for (int e = 0; e < Dm; e++) a += ffin[e] * fc1W[(size_t)j * Dm + e];
      ff1[j] = fmaxf(a, 0.f);
    }
    __syncthreads();
    float o = fc2b[t];
    for (int e = 0; e < Hm; e++) o += ff1[e] * fc2W[(size_t)t * Hm + e];
    float res = h + o;
    __syncthreads();
    slots_out[((size_t)b * SP + s) * Dm + t] = res;
    out_slots[((size_t)(b * Sm + s)) * Dm + t] = res;
  }
}

// ---------------------------------------------------------------------------
extern "C" void kernel_launch(void* const* d_in, const int* in_sizes, int n_in,
                              void* d_out, int out_size, void* d_ws, size_t ws_size,
                              hipStream_t stream) {
  const float* inputs    = (const float*)d_in[0];
  const float* slots_ini = (const float*)d_in[1];
  const float* slots_mu  = (const float*)d_in[2];
  const float* slots_ls  = (const float*)d_in[3];
  const float* Wq   = (const float*)d_in[4];
  const float* Wk   = (const float*)d_in[5];
  const float* Wv   = (const float*)d_in[6];
  const float* gWih = (const float*)d_in[7];
  const float* gWhh = (const float*)d_in[8];
  const float* gbih = (const float*)d_in[9];
  const float* gbhh = (const float*)d_in[10];
  const float* fc1W = (const float*)d_in[11];
  const float* fc1b = (const float*)d_in[12];
  const float* fc2W = (const float*)d_in[13];
  const float* fc2b = (const float*)d_in[14];
  const float* lnig = (const float*)d_in[15];
  const float* lnib = (const float*)d_in[16];
  const float* lnsg = (const float*)d_in[17];
  const float* lnsb = (const float*)d_in[18];
  const float* lnfg = (const float*)d_in[19];
  const float* lnfb = (const float*)d_in[20];
  const float* wiW  = (const float*)d_in[21];
  const float* wib  = (const float*)d_in[22];
  const float* wsW  = (const float*)d_in[23];
  const float* wsb  = (const float*)d_in[24];

  char* ws = (char*)d_ws;
  size_t off = 0;
  auto alloc = [&](size_t bytes) -> char* {
    char* p = ws + off;
    off += (bytes + 255) & ~(size_t)255;
    return p;
  };
  h16*   kh    = (h16*)alloc((size_t)NT * Dm * sizeof(h16));        // 32 MB row-major
  h16*   vhT   = (h16*)alloc((size_t)NT * Dm * sizeof(h16));        // 32 MB [b][d][n]
  h16*   attnT = (h16*)alloc((size_t)NT * SP * sizeof(h16));        // 4 MB  [b][s][n]
  float* C     = (float*)alloc((size_t)NT * SP * sizeof(float));    // 8 MB
  float* k2    = (float*)alloc((size_t)NT * sizeof(float));
  float* wlog  = (float*)alloc((size_t)NT * sizeof(float));
  float* log_a = (float*)alloc((size_t)NT * sizeof(float));
  float* u     = (float*)alloc((size_t)NT * sizeof(float));
  float* vdual = (float*)alloc((size_t)Bm * SP * sizeof(float));
  float* log_b = (float*)alloc((size_t)Bm * SP * sizeof(float));
  float* q2    = (float*)alloc((size_t)Bm * SP * sizeof(float));
  h16*   qh    = (h16*)alloc((size_t)Bm * 16 * Dm * sizeof(h16));
  float* slots = (float*)alloc((size_t)Bm * SP * Dm * sizeof(float));
  float* upd   = (float*)alloc((size_t)Bm * SP * Dm * sizeof(float));
  if (off > ws_size) return;  // scratch too small; avoid corruption

  float* out_slots = (float*)d_out;
  float* attn_out  = (float*)d_out + Bm * Sm * Dm;

  k_init_slots<<<(Bm * Sm * Dm + 255) / 256, 256, 0, stream>>>(slots_ini, slots_mu, slots_ls, slots);
  k_ln_proj<<<NT / 128, 256, 0, stream>>>(inputs, Wk, Wv, lnig, lnib, wiW, wib, kh, vhT, wlog);
  k_k2<<<NT / 256, 256, 0, stream>>>(kh, k2);
  k_amarg<<<Bm, 256, 0, stream>>>(wlog, log_a);

  for (int it = 0; it < 3; it++) {
    k_slotsnorm<<<Bm, 64, 0, stream>>>(slots, lnsg, lnsb, wsW, wsb, Wq, log_b, q2, qh, vdual);
    k_cdist<<<Bm * (Nn / 16), 32, 0, stream>>>(kh, qh, k2, q2, C);
    for (int ms = 0; ms < 4; ms++) {          // MESH iterations
      for (int sh = 0; sh < 5; sh++) {        // inner Sinkhorn (warm-started)
        k_u<<<NT / 256, 256, 0, stream>>>(C, vdual, log_a, u);
        k_v<<<Bm, 256, 0, stream>>>(C, u, log_b, vdual);
      }
      k_meshgrad<<<NT / 256, 256, 0, stream>>>(C, u, vdual);
    }
    for (int sh = 0; sh < 5; sh++) {          // final Sinkhorn on updated C
      k_u<<<NT / 256, 256, 0, stream>>>(C, vdual, log_a, u);
      k_v<<<Bm, 256, 0, stream>>>(C, u, log_b, vdual);
    }
    k_attn<<<NT / 256, 256, 0, stream>>>(C, u, vdual, attnT, attn_out);
    k_updates<<<Bm * 4, 256, 0, stream>>>(attnT, vhT, upd);
    k_gru_ff<<<Bm, 64, 0, stream>>>(upd, slots, gWih, gWhh, gbih, gbhh,
                                    fc1W, fc1b, fc2W, fc2b, lnfg, lnfb,
                                    slots, out_slots);
  }
}
